// MixtureOfExperts_56796647523001
// MI455X (gfx1250) — matrix operand order repaired and checked by reading, not compiled, not run on hardware
//
#include <hip/hip_runtime.h>
#include <hip/hip_bf16.h>
#include <math.h>

// ---------------------------------------------------------------------------
// MoE forward for MI455X (gfx1250, wave32, WMMA 16x16x32 bf16)
//   B=16384, D=1024, H=512, O=256, E=8
// Per-wave register tile: 32(M) x 64(N)  -> 8 WMMAs per 12 b128 loads
// Block: 8 waves = 4(M) x 2(N) wave grid -> 128 x 128 block tile
// ---------------------------------------------------------------------------

typedef __bf16 v16bf __attribute__((ext_vector_type(16)));
typedef float  v8f   __attribute__((ext_vector_type(8)));

union FragBF {
  v16bf v;
  uint4 u4[2];   // two b128 loads fill one 16-element bf16 fragment
};

__device__ __forceinline__ unsigned short f2bf(float f) {
  unsigned int u = __float_as_uint(f);
  u += 0x7FFFu + ((u >> 16) & 1u);   // round-to-nearest-even
  return (unsigned short)(u >> 16);
}
__device__ __forceinline__ float bf2f(unsigned short h) {
  return __uint_as_float(((unsigned int)h) << 16);
}

// ---------------------------------------------------------------------------
// Elementwise f32 -> bf16
// ---------------------------------------------------------------------------
__global__ void k_f32_to_bf16(const float* __restrict__ src,
                              unsigned short* __restrict__ dst, int n) {
  int i = blockIdx.x * blockDim.x + threadIdx.x;
  if (i < n) dst[i] = f2bf(src[i]);
}

// ---------------------------------------------------------------------------
// Transpose+convert weights: src f32 [K,N] (slice z) -> dst bf16 [N,K]
// ---------------------------------------------------------------------------
__global__ void k_transpose_bf16(const float* __restrict__ src,
                                 unsigned short* __restrict__ dst,
                                 int K, int N) {
  long long slice = blockIdx.z;
  const float* s = src + slice * (long long)K * N;
  unsigned short* d = dst + slice * (long long)K * N;
  int i = blockIdx.x * blockDim.x + threadIdx.x;
  if (i < K * N) {
    int k = i / N, n = i - k * N;
    d[(long long)n * K + k] = f2bf(s[i]);
  }
}

// ---------------------------------------------------------------------------
// Shared 32x64-per-wave WMMA core.
//   A bf16 [M,K] row-major, Wt bf16 [N,K] row-major (pre-transposed weights).
//   Computes acc[2][4] (two 16-row groups x four 16-col groups).
// ---------------------------------------------------------------------------
#define WMMA_TILE_LOOP(A_, Wt_, Kv_, acc_)                                        \
  const int lmod  = lane & 15;                                                    \
  const int lhalf = lane >> 4;                                                    \
  const unsigned short* arow0 = (A_) + (size_t)(mtile + lmod) * (Kv_) + lhalf * 16;      \
  const unsigned short* arow1 = (A_) + (size_t)(mtile + 16 + lmod) * (Kv_) + lhalf * 16; \
  const unsigned short* wrow0 = (Wt_) + (size_t)(ntile + 0 * 16 + lmod) * (Kv_) + lhalf * 16; \
  const unsigned short* wrow1 = (Wt_) + (size_t)(ntile + 1 * 16 + lmod) * (Kv_) + lhalf * 16; \
  const unsigned short* wrow2 = (Wt_) + (size_t)(ntile + 2 * 16 + lmod) * (Kv_) + lhalf * 16; \
  const unsigned short* wrow3 = (Wt_) + (size_t)(ntile + 3 * 16 + lmod) * (Kv_) + lhalf * 16; \
  for (int k = 0; k < (Kv_); k += 32) {                                           \
    FragBF a0, a1, b0, b1, b2, b3;                                                \
    a0.u4[0] = *(const uint4*)(arow0 + k);                                        \
    a0.u4[1] = *(const uint4*)(arow0 + k + 8);                                    \
    a1.u4[0] = *(const uint4*)(arow1 + k);                                        \
    a1.u4[1] = *(const uint4*)(arow1 + k + 8);                                    \
    b0.u4[0] = *(const uint4*)(wrow0 + k);                                        \
    b0.u4[1] = *(const uint4*)(wrow0 + k + 8);                                    \
    b1.u4[0] = *(const uint4*)(wrow1 + k);                                        \
    b1.u4[1] = *(const uint4*)(wrow1 + k + 8);                                    \
    b2.u4[0] = *(const uint4*)(wrow2 + k);                                        \
    b2.u4[1] = *(const uint4*)(wrow2 + k + 8);                                    \
    b3.u4[0] = *(const uint4*)(wrow3 + k);                                        \
    b3.u4[1] = *(const uint4*)(wrow3 + k + 8);                                    \
    __builtin_prefetch(arow0 + k + 256, 0, 3);  /* global_prefetch_b8 */          \
    acc_[0][0] = __builtin_amdgcn_wmma_f32_16x16x32_bf16(false, a0.v, false, b0.v, (short)0, acc_[0][0], false, false); \
    acc_[0][1] = __builtin_amdgcn_wmma_f32_16x16x32_bf16(false, a0.v, false, b1.v, (short)0, acc_[0][1], false, false); \
    acc_[0][2] = __builtin_amdgcn_wmma_f32_16x16x32_bf16(false, a0.v, false, b2.v, (short)0, acc_[0][2], false, false); \
    acc_[0][3] = __builtin_amdgcn_wmma_f32_16x16x32_bf16(false, a0.v, false, b3.v, (short)0, acc_[0][3], false, false); \
    acc_[1][0] = __builtin_amdgcn_wmma_f32_16x16x32_bf16(false, a1.v, false, b0.v, (short)0, acc_[1][0], false, false); \
    acc_[1][1] = __builtin_amdgcn_wmma_f32_16x16x32_bf16(false, a1.v, false, b1.v, (short)0, acc_[1][1], false, false); \
    acc_[1][2] = __builtin_amdgcn_wmma_f32_16x16x32_bf16(false, a1.v, false, b2.v, (short)0, acc_[1][2], false, false); \
    acc_[1][3] = __builtin_amdgcn_wmma_f32_16x16x32_bf16(false, a1.v, false, b3.v, (short)0, acc_[1][3], false, false); \
  }

// ---------------------------------------------------------------------------
// C_bf16[M,N] = relu(A_bf16[M,K] @ W + bias),  W given transposed: Wt[N,K]
// ---------------------------------------------------------------------------
__global__ void __launch_bounds__(256)
k_gemm_bf16_bias_relu(const unsigned short* __restrict__ A,
                      const unsigned short* __restrict__ Wt,
                      const float* __restrict__ bias,
                      unsigned short* __restrict__ C,
                      int M, int N, int K) {
  const int lane  = threadIdx.x & 31;
  const int wave  = threadIdx.x >> 5;
  const int mtile = (blockIdx.x * 4 + (wave & 3)) * 32;
  const int ntile = (blockIdx.y * 2 + (wave >> 2)) * 64;
  if (mtile >= M) return;

  v8f acc[2][4] = {};
  WMMA_TILE_LOOP(A, Wt, K, acc)

  // C/D layout: VGPR r -> M = i*16 + lhalf*8 + r, N = ntile + t*16 + lmod
#pragma unroll
  for (int i = 0; i < 2; ++i) {
#pragma unroll
    for (int t = 0; t < 4; ++t) {
      const int n = ntile + t * 16 + lmod;
      const float bv = bias[n];
#pragma unroll
      for (int r = 0; r < 8; ++r) {
        const int m = mtile + i * 16 + lhalf * 8 + r;
        float v = acc[i][t][r] + bv;
        v = v > 0.0f ? v : 0.0f;
        C[(size_t)m * N + n] = f2bf(v);
      }
    }
  }
}

// ---------------------------------------------------------------------------
// out_f32[M,N] (+)= gates[m,e] * (A_bf16[M,K] @ W + bias), Wt[N,K]
// ---------------------------------------------------------------------------
__global__ void __launch_bounds__(256)
k_gemm_bf16_gated_out(const unsigned short* __restrict__ A,
                      const unsigned short* __restrict__ Wt,
                      const float* __restrict__ bias,
                      const float* __restrict__ gates,   // [M, 8]
                      float* __restrict__ out,
                      int M, int N, int K, int e, int accumulate) {
  const int lane  = threadIdx.x & 31;
  const int wave  = threadIdx.x >> 5;
  const int mtile = (blockIdx.x * 4 + (wave & 3)) * 32;
  const int ntile = (blockIdx.y * 2 + (wave >> 2)) * 64;
  if (mtile >= M) return;

  v8f acc[2][4] = {};
  WMMA_TILE_LOOP(A, Wt, K, acc)

#pragma unroll
  for (int i = 0; i < 2; ++i) {
#pragma unroll
    for (int r = 0; r < 8; ++r) {
      const int m = mtile + i * 16 + lhalf * 8 + r;
      const float g = gates[(size_t)m * 8 + e];
#pragma unroll
      for (int t = 0; t < 4; ++t) {
        const int n = ntile + t * 16 + lmod;
        float v = (acc[i][t][r] + bias[n]) * g;
        float* p = out + (size_t)m * N + n;
        if (accumulate) *p = *p + v; else *p = v;
      }
    }
  }
}

// ---------------------------------------------------------------------------
// Gating head: logits = g2 @ gw3 + gb3, softmax over E=8.  One thread per row.
// ---------------------------------------------------------------------------
__global__ void k_gating_softmax(const unsigned short* __restrict__ g2, // [B,128] bf16
                                 const float* __restrict__ gw3,         // [128,8]
                                 const float* __restrict__ gb3,         // [8]
                                 float* __restrict__ gates, int B) {
  int b = blockIdx.x * blockDim.x + threadIdx.x;
  if (b >= B) return;
  float logit[8];
#pragma unroll
  for (int e = 0; e < 8; ++e) logit[e] = gb3[e];
  const unsigned short* row = g2 + (size_t)b * 128;
  for (int j = 0; j < 128; ++j) {
    float xv = bf2f(row[j]);
#pragma unroll
    for (int e = 0; e < 8; ++e) logit[e] += xv * gw3[j * 8 + e];
  }
  float mx = logit[0];
#pragma unroll
  for (int e = 1; e < 8; ++e) mx = fmaxf(mx, logit[e]);
  float s = 0.0f;
#pragma unroll
  for (int e = 0; e < 8; ++e) { float p = __expf(logit[e] - mx); logit[e] = p; s += p; }
  float inv = 1.0f / s;
#pragma unroll
  for (int e = 0; e < 8; ++e) gates[(size_t)b * 8 + e] = logit[e] * inv;
}

// ---------------------------------------------------------------------------
// Host-side orchestration
// ---------------------------------------------------------------------------
extern "C" void kernel_launch(void* const* d_in, const int* in_sizes, int n_in,
                              void* d_out, int out_size, void* d_ws, size_t ws_size,
                              hipStream_t stream) {
  const float* x   = (const float*)d_in[0];
  const float* gw1 = (const float*)d_in[1];
  const float* gb1 = (const float*)d_in[2];
  const float* gw2 = (const float*)d_in[3];
  const float* gb2 = (const float*)d_in[4];
  const float* gw3 = (const float*)d_in[5];
  const float* gb3 = (const float*)d_in[6];
  const float* ew1 = (const float*)d_in[7];
  const float* eb1 = (const float*)d_in[8];
  const float* ew2 = (const float*)d_in[9];
  const float* eb2 = (const float*)d_in[10];
  const float* ew3 = (const float*)d_in[11];
  const float* eb3 = (const float*)d_in[12];
  float* out = (float*)d_out;
  (void)in_sizes; (void)n_in; (void)out_size; (void)ws_size;

  const int B = 16384, D = 1024, H = 512, O = 256, E = 8;

  char* ws = (char*)d_ws;
  size_t off = 0;
  auto alloc = [&](size_t bytes) -> void* {
    void* p = ws + off;
    off = (off + bytes + 255) & ~(size_t)255;
    return p;
  };

  unsigned short* xb    = (unsigned short*)alloc((size_t)B * D * 2);       // 32 MB
  unsigned short* g1    = (unsigned short*)alloc((size_t)B * 256 * 2);     //  8 MB
  unsigned short* g2    = (unsigned short*)alloc((size_t)B * 128 * 2);     //  4 MB
  float*          gates = (float*)         alloc((size_t)B * E * 4);       // 0.5 MB
  unsigned short* gw1t  = (unsigned short*)alloc((size_t)256 * D * 2);
  unsigned short* gw2t  = (unsigned short*)alloc((size_t)128 * 256 * 2);
  unsigned short* ew1t  = (unsigned short*)alloc((size_t)E * D * H * 2);   //  8 MB
  unsigned short* ew2t  = (unsigned short*)alloc((size_t)E * H * H * 2);   //  4 MB
  unsigned short* ew3t  = (unsigned short*)alloc((size_t)E * H * O * 2);   //  2 MB
  unsigned short* h1    = (unsigned short*)alloc((size_t)B * H * 2);       // 16 MB
  unsigned short* h2    = (unsigned short*)alloc((size_t)B * H * 2);       // 16 MB

  // --- precision conversion / weight transposition ---
  k_f32_to_bf16<<<(B * D + 255) / 256, 256, 0, stream>>>(x, xb, B * D);
  k_transpose_bf16<<<dim3((D * 256 + 255) / 256, 1, 1), 256, 0, stream>>>(gw1, gw1t, D, 256);
  k_transpose_bf16<<<dim3((256 * 128 + 255) / 256, 1, 1), 256, 0, stream>>>(gw2, gw2t, 256, 128);
  k_transpose_bf16<<<dim3((D * H + 255) / 256, 1, E), 256, 0, stream>>>(ew1, ew1t, D, H);
  k_transpose_bf16<<<dim3((H * H + 255) / 256, 1, E), 256, 0, stream>>>(ew2, ew2t, H, H);
  k_transpose_bf16<<<dim3((H * O + 255) / 256, 1, E), 256, 0, stream>>>(ew3, ew3t, H, O);

  dim3 blk(256);

  // --- gating network (block tile 128x128) ---
  k_gemm_bf16_bias_relu<<<dim3(B / 128, 256 / 128), blk, 0, stream>>>(xb, gw1t, gb1, g1, B, 256, D);
  k_gemm_bf16_bias_relu<<<dim3(B / 128, 128 / 128), blk, 0, stream>>>(g1, gw2t, gb2, g2, B, 128, 256);
  k_gating_softmax<<<(B + 255) / 256, 256, 0, stream>>>(g2, gw3, gb3, gates, B);

  // --- experts (ping-pong h1/h2, gate-weighted accumulation into out) ---
  for (int e = 0; e < E; ++e) {
    k_gemm_bf16_bias_relu<<<dim3(B / 128, H / 128), blk, 0, stream>>>(
        xb, ew1t + (size_t)e * D * H, eb1 + (size_t)e * H, h1, B, H, D);
    k_gemm_bf16_bias_relu<<<dim3(B / 128, H / 128), blk, 0, stream>>>(
        h1, ew2t + (size_t)e * H * H, eb2 + (size_t)e * H, h2, B, H, H);
    k_gemm_bf16_gated_out<<<dim3(B / 128, O / 128), blk, 0, stream>>>(
        h2, ew3t + (size_t)e * H * O, eb3 + (size_t)e * O, gates, out, B, O, H, e, e > 0);
  }
}
